// DGCNN_89223650607921
// MI455X (gfx1250) — matrix operand order repaired and checked by reading, not compiled, not run on hardware
//
#include <hip/hip_runtime.h>

// ---------------------------------------------------------------------------
// DGCNN forward for MI455X (gfx1250, wave32).
//  - all contractions via v_wmma_f32_16x16x32_bf16 (f32 accumulate)
//  - position-major bf16 activations -> vectorized fragment loads
//  - dense GEMMs stage activations into LDS with the Tensor Data Mover,
//    double-buffered (issue DMA i+1, s_wait_tensorcnt 1, compute on i)
// ---------------------------------------------------------------------------

#define BATCH 8
#define NPTS  4096
#define KNN   20
#define BN    (BATCH * NPTS)            // 32768 points
#define P1LL  ((long long)BN * KNN)     // 655360 edge positions

typedef __bf16 bf16;
typedef __bf16 v16bf __attribute__((ext_vector_type(16)));
typedef __bf16 v8bf  __attribute__((ext_vector_type(8)));
typedef float  v8f   __attribute__((ext_vector_type(8)));
typedef unsigned int u32;
typedef u32 u32x4 __attribute__((ext_vector_type(4)));
typedef int i32x4 __attribute__((ext_vector_type(4)));
typedef int i32x8 __attribute__((ext_vector_type(8)));

static __device__ __forceinline__ v8f wmma_bf16(v16bf a, v16bf b, v8f c) {
  return __builtin_amdgcn_wmma_f32_16x16x32_bf16(false, a, false, b, (short)0, c,
                                                 false, false);
}

static __device__ __forceinline__ v16bf cat8(v8bf lo, v8bf hi) {
  return __builtin_shufflevector(lo, hi, 0, 1, 2, 3, 4, 5, 6, 7, 8, 9, 10, 11,
                                 12, 13, 14, 15);
}

// Branchless sorted-descending top-20 insert (v_cndmask chains, no EXEC churn).
static __device__ __forceinline__ void insert20(float (&bv)[KNN], int (&bi)[KNN],
                                                float d, int m) {
  float cv = d;
  int ci = m;
#pragma unroll
  for (int j = 0; j < KNN; ++j) {
    bool gt = cv > bv[j];
    float nv = gt ? cv : bv[j];
    int ni = gt ? ci : bi[j];
    float ov = gt ? bv[j] : cv;
    int oi = gt ? bi[j] : ci;
    bv[j] = nv; bi[j] = ni; cv = ov; ci = oi;
  }
}

// ---------------------------------------------------------------------------
// Tensor Data Mover: 2-D tile load (tile_y rows x tile_x bf16 elems) from a
// position-major matrix with row stride `stride` elems into LDS at lds_off.
// D# packing per CDNA5 ISA (group0: count/lds/global/type, group1: dims).
// ---------------------------------------------------------------------------
static __device__ __forceinline__ void tdm_load_2d(unsigned lds_off,
                                                   const bf16* gptr, int tile_x,
                                                   int tile_y, int stride) {
  unsigned long long ga = (unsigned long long)(const void*)gptr;
  u32x4 g0;
  g0[0] = 1u;                                        // count=1, user mode
  g0[1] = lds_off;                                   // lds_addr (bytes)
  g0[2] = (u32)(ga & 0xffffffffu);                   // global_addr[31:0]
  g0[3] = (u32)((ga >> 32) & 0x1ffffffu) | (2u << 30); // addr[56:32] | type=2
  const u32 td0 = 1u << 30, td1 = 1u << 30;          // huge: no OOB clipping
  i32x8 g1;
  g1[0] = (int)(1u << 16);                           // data_size=1 (2 bytes)
  g1[1] = (int)((td0 & 0xffffu) << 16);              // [31:16]=tensor_dim0 lo
  g1[2] = (int)((td0 >> 16) | ((td1 & 0xffffu) << 16));
  g1[3] = (int)((td1 >> 16) | ((u32)tile_x << 16));  // [31:16]=tile_dim0
  g1[4] = (int)((u32)tile_y);                        // tile_dim1, tile_dim2=0
  g1[5] = (int)(u32)stride;                          // tensor_dim0_stride lo
  g1[6] = 0;                                         // stride hi, dim1_stride=0
  g1[7] = 0;
  i32x4 gz = {0, 0, 0, 0};
#if defined(__clang_major__) && (__clang_major__ >= 23)
  i32x8 gz8 = {0, 0, 0, 0, 0, 0, 0, 0};
  __builtin_amdgcn_tensor_load_to_lds(g0, g1, gz, gz, gz8, 0);
#else
  __builtin_amdgcn_tensor_load_to_lds(g0, g1, gz, gz, 0);
#endif
}

// ---------------------------------------------------------------------------
// Weight pre-swizzle: Wsw[((mt*nkt+kt)*32 + lane)*16 + e] in contiguous-k A-fragment
// order (lane -> row M=lane%16, K = 2*base + e, base=0/8).
// ---------------------------------------------------------------------------
__global__ void prep_w(const float* __restrict__ W, bf16* __restrict__ Wsw,
                       int Cout, int Cin, int Mtiles, int nkt) {
  long long i = (long long)blockIdx.x * blockDim.x + threadIdx.x;
  long long tot = (long long)Mtiles * nkt * 512;
  if (i >= tot) return;
  int e = (int)(i & 15);
  int lane = (int)((i >> 4) & 31);
  long long rem = i >> 9;
  int kt = (int)(rem % nkt);
  int mt = (int)(rem / nkt);
  int m = mt * 16 + (lane & 15);
  int base = (lane < 16) ? 0 : 8;
  int K = (base << 1) + e;
  int cc = (kt << 5) + K;
  Wsw[i] = (m < Cout && cc < Cin) ? (bf16)W[(long long)m * Cin + cc] : (bf16)0.f;
}

// ---------------------------------------------------------------------------
// GEMM: Y[m][p] = sum_c W[m][c] * X[c at position p]
//  MODE 0: dense, X position-major [P][Cin] (Cin % 64 == 0), double-buffered
//          TDM -> LDS staging
//  MODE 1: edge gather, src [BN][64], Cin=128 (diff | center), vector loads
//  MODE 2: edge gather, src [BN][16] (9 real ch, zero-padded), Cin=18, scalar
// ---------------------------------------------------------------------------
template <int MODE>
__global__ void __launch_bounds__(128)
gemm_wmma(const bf16* __restrict__ Wsw, const bf16* __restrict__ X,
          const int* __restrict__ idx, float* __restrict__ Y, int Cout, int Cin,
          long long P) {
  const int lane = threadIdx.x & 31;
  const int wave = threadIdx.x >> 5;
  const int mt = blockIdx.x;
  const long long pt = (long long)blockIdx.y * 4 + wave;
  if (pt * 16 >= P) return;                    // uniform across the block

  const int p = (int)(pt * 16) + (lane & 15);  // this lane's output column
  const int abase = (lane < 16) ? 0 : 8;
  const int kbase = (lane >> 4) << 4;
  const int nkt = (Cin + 31) >> 5;

  int nglob = 0, jglob = 0;
  if (MODE != 0) {
    int pn = p / KNN;
    nglob = pn;
    jglob = ((pn >> 12) << 12) + idx[p];
  }

  v8f acc = {};

  if (MODE == 0) {
    // double-buffered LDS staging: 2 x (64 positions x 64 channels)
    __shared__ bf16 tile[2][64 * 64];
    const int prel = (wave << 4) + (lane & 15);
    const bf16* strip = X + (long long)blockIdx.y * 64 * Cin;
    if (wave == 0)                              // prologue: DMA chunk 0
      tdm_load_2d((unsigned)(unsigned long long)(void*)&tile[0][0], strip, 64,
                  64, Cin);
    int buf = 0;
    for (int c0 = 0; c0 < Cin; c0 += 64) {
      if (wave == 0) {
        bool more = (c0 + 64) < Cin;
        if (more) {                             // DMA chunk i+1 into other buf
          tdm_load_2d((unsigned)(unsigned long long)(void*)&tile[buf ^ 1][0],
                      strip + c0 + 64, 64, 64, Cin);
          // TDM ops from one wave complete in order: tensorcnt<=1 means the
          // older DMA (chunk i) has landed while chunk i+1 still streams.
          __builtin_amdgcn_s_wait_tensorcnt(1);
        } else {
          __builtin_amdgcn_s_wait_tensorcnt(0); // epilogue: drain
        }
      }
      __syncthreads();                          // tile[buf] ready for all waves
#pragma unroll
      for (int kk = 0; kk < 2; ++kk) {
        int kt = (c0 >> 5) + kk;
        v16bf a = *(const v16bf*)(Wsw + (((long long)mt * nkt + kt) << 9) +
                                  (lane << 4));
        const bf16* bp = &tile[buf][prel * 64 + (kk << 5) + kbase];
        v16bf bfr = cat8(*(const v8bf*)bp, *(const v8bf*)(bp + 8));
        acc = wmma_bf16(a, bfr, acc);
      }
      __syncthreads();                          // done reading tile[buf]
      buf ^= 1;
    }
  } else if (MODE == 1) {                      // Csrc = 64, Cin = 128
    const bf16* rj = X + (long long)jglob * 64;
    const bf16* rn = X + (long long)nglob * 64;
    for (int kt = 0; kt < nkt; ++kt) {
      int cc0 = (kt << 5) + kbase;
      v8bf lo, hi;
      if (cc0 < 64) {                          // feat - center
        v8bf jl = *(const v8bf*)(rj + cc0), jh = *(const v8bf*)(rj + cc0 + 8);
        v8bf nl = *(const v8bf*)(rn + cc0), nh = *(const v8bf*)(rn + cc0 + 8);
#pragma unroll
        for (int e = 0; e < 8; ++e) {
          lo[e] = (bf16)((float)jl[e] - (float)nl[e]);
          hi[e] = (bf16)((float)jh[e] - (float)nh[e]);
        }
      } else {                                 // center
        lo = *(const v8bf*)(rn + cc0 - 64);
        hi = *(const v8bf*)(rn + cc0 - 64 + 8);
      }
      v16bf a = *(const v16bf*)(Wsw + (((long long)mt * nkt + kt) << 9) +
                                (lane << 4));
      acc = wmma_bf16(a, cat8(lo, hi), acc);
    }
  } else {                                     // MODE 2: Csrc = 9, Cin = 18
    const bf16* rj = X + (long long)jglob * 16;
    const bf16* rn = X + (long long)nglob * 16;
    v16bf a = *(const v16bf*)(Wsw + (((long long)mt * nkt + 0) << 9) +
                              (lane << 4));
    v16bf bfr;
#pragma unroll
    for (int e = 0; e < 16; ++e) {
      int cc = kbase + e;
      float v;
      if (cc < 9)       v = (float)rj[cc] - (float)rn[cc];
      else if (cc < 18) v = (float)rn[cc - 9];
      else              v = 0.f;
      bfr[e] = (bf16)v;
    }
    acc = wmma_bf16(a, bfr, acc);
  }

  const int mrow = mt * 16 + abase;
#pragma unroll
  for (int v = 0; v < 8; ++v) {
    int mm = mrow + v;
    if (mm < Cout) Y[(long long)mm * P + p] = acc[v];
  }
}

// ---------------------------------------------------------------------------
// BatchNorm (training-mode, biased var) per channel; Y is channel-major f32.
// ---------------------------------------------------------------------------
__global__ void colstats(const float* __restrict__ Y, long long P,
                         float* __restrict__ mean, float* __restrict__ var) {
  int c = blockIdx.x;
  const float* row = Y + (long long)c * P;
  float s = 0.f, s2 = 0.f;
  for (long long p = threadIdx.x; p < P; p += blockDim.x) {
    float v = row[p];
    s += v; s2 += v * v;
  }
  __shared__ float ss[256], sq[256];
  ss[threadIdx.x] = s; sq[threadIdx.x] = s2;
  __syncthreads();
  for (int o = 128; o > 0; o >>= 1) {
    if ((int)threadIdx.x < o) {
      ss[threadIdx.x] += ss[threadIdx.x + o];
      sq[threadIdx.x] += sq[threadIdx.x + o];
    }
    __syncthreads();
  }
  if (threadIdx.x == 0) {
    float m = ss[0] / (float)P;
    mean[c] = m;
    var[c] = sq[0] / (float)P - m * m;
  }
}

// normalize + LeakyReLU, write position-major bf16 [P][Cpad]
__global__ void bn_lrelu_bf16(const float* __restrict__ Y,
                              const float* __restrict__ mean,
                              const float* __restrict__ var,
                              const float* __restrict__ g,
                              const float* __restrict__ b,
                              bf16* __restrict__ out, long long total,
                              long long P, int Cpad) {
  long long i = (long long)blockIdx.x * blockDim.x + threadIdx.x;
  if (i >= total) return;
  long long c = i / P, p = i % P;
  float x = (Y[i] - mean[c]) * rsqrtf(var[c] + 1e-5f) * g[c] + b[c];
  out[p * Cpad + c] = (bf16)(x >= 0.f ? x : 0.2f * x);
}

// ---------------------------------------------------------------------------
// Elementwise / reshuffle kernels (all position-major bf16)
// ---------------------------------------------------------------------------
__global__ void make_xin(const float* __restrict__ x, bf16* __restrict__ xin) {
  long long i = (long long)blockIdx.x * blockDim.x + threadIdx.x; // [BN][16]
  if (i >= (long long)BN * 16) return;
  long long pn = i >> 4, c = i & 15;
  long long b = pn >> 12, n = pn & 4095;
  xin[i] = (c < 9) ? (bf16)x[(b * 9 + c) * NPTS + n] : (bf16)0.f;
}

__global__ void kmax20(const bf16* __restrict__ in, bf16* __restrict__ out) {
  long long i = (long long)blockIdx.x * blockDim.x + threadIdx.x; // [BN][64]
  if (i >= (long long)BN * 64) return;
  long long pn = i >> 6, c = i & 63;
  float m = -1e30f;
#pragma unroll
  for (int k = 0; k < KNN; ++k)
    m = fmaxf(m, (float)in[(((long long)pn * KNN + k) << 6) + c]);
  out[i] = (bf16)m;
}

__global__ void xnorm64(const bf16* __restrict__ xb, float* __restrict__ xx) {
  int p = blockIdx.x * blockDim.x + threadIdx.x;
  if (p >= BN) return;
  const bf16* r = xb + (long long)p * 64;
  float s = 0.f;
#pragma unroll 8
  for (int c = 0; c < 64; ++c) {
    float v = (float)r[c];
    s += v * v;
  }
  xx[p] = s;
}

// copy 64-channel position-major block into column slice of [P][Cdst]
__global__ void copy_cols(const bf16* __restrict__ s, bf16* __restrict__ d,
                          int Cdst, int coff, long long P) {
  long long i = (long long)blockIdx.x * blockDim.x + threadIdx.x;
  if (i >= P * 64) return;
  long long p = i >> 6, c = i & 63;
  d[p * Cdst + coff + c] = s[i];
}

__global__ void gmax_kernel(const bf16* __restrict__ e, bf16* __restrict__ gm) {
  int c = blockIdx.x, b = blockIdx.y;          // e: [BN][1024]
  float m = -1e30f;
  for (int n = threadIdx.x; n < NPTS; n += 256)
    m = fmaxf(m, (float)e[((long long)b * NPTS + n) * 1024 + c]);
  __shared__ float sm[256];
  sm[threadIdx.x] = m;
  __syncthreads();
  for (int o = 128; o > 0; o >>= 1) {
    if ((int)threadIdx.x < o)
      sm[threadIdx.x] = fmaxf(sm[threadIdx.x], sm[threadIdx.x + o]);
    __syncthreads();
  }
  if (threadIdx.x == 0) gm[c * BATCH + b] = (bf16)sm[0];
}

__global__ void bcast_gm(const bf16* __restrict__ gm, bf16* __restrict__ dst) {
  long long i = (long long)blockIdx.x * blockDim.x + threadIdx.x; // [BN][1024]
  if (i >= (long long)BN * 1024) return;
  long long p = i >> 10, c = i & 1023;
  dst[p * 1216 + c] = gm[c * BATCH + (p >> 12)];
}

__global__ void out_xpose(const float* __restrict__ Yc, float* __restrict__ out) {
  long long i = (long long)blockIdx.x * blockDim.x + threadIdx.x; // [BN][17]
  if (i >= (long long)17 * BN) return;
  long long pn = i / 17, o = i % 17;
  out[i] = Yc[o * (long long)BN + pn];
}

// ---------------------------------------------------------------------------
// kNN on the 3 coordinate channels (x[:,6:9,:])
// ---------------------------------------------------------------------------
__global__ void knn3(const float* __restrict__ x, int* __restrict__ idx) {
  int t = blockIdx.x * blockDim.x + threadIdx.x;
  if (t >= BN) return;
  int b = t >> 12, n = t & 4095;
  const float* xb = x + (long long)b * 9 * NPTS;
  float qx = xb[6 * NPTS + n], qy = xb[7 * NPTS + n], qz = xb[8 * NPTS + n];
  float bv[KNN]; int bi[KNN];
#pragma unroll
  for (int j = 0; j < KNN; ++j) { bv[j] = -1e30f; bi[j] = 0; }
  for (int m = 0; m < NPTS; ++m) {
    float dx = xb[6 * NPTS + m] - qx;
    float dy = xb[7 * NPTS + m] - qy;
    float dz = xb[8 * NPTS + m] - qz;
    float d = -(dx * dx + dy * dy + dz * dz);
    if (d > bv[KNN - 1]) insert20(bv, bi, d, m);
  }
#pragma unroll
  for (int j = 0; j < KNN; ++j) idx[(long long)t * KNN + j] = bi[j];
}

// ---------------------------------------------------------------------------
// kNN on 64 channels via WMMA Gram tiles (d = 2*inner - |q|^2 - |m|^2).
// D tile: M = neighbor, N = query. Lane owns query lane%16 + 8 neighbor rows;
// lanes L / L+16 merge their sorted top-20 through LDS.
// ---------------------------------------------------------------------------
__global__ void __launch_bounds__(32)
knn64(const bf16* __restrict__ xb, const float* __restrict__ xx,
      int* __restrict__ idx) {
  const int lane = threadIdx.x;
  const int q0 = blockIdx.x * 16;
  const int b = blockIdx.y;
  const long long base = (long long)b * NPTS;

  const int q = q0 + (lane & 15);
  const int kb = (lane >> 4) << 4;
  const bf16* rq = xb + (base + q) * 64;
  v16bf bf0 = cat8(*(const v8bf*)(rq + kb), *(const v8bf*)(rq + kb + 8));
  v16bf bf1 = cat8(*(const v8bf*)(rq + 32 + kb), *(const v8bf*)(rq + 32 + kb + 8));
  const float xxq = xx[base + q];

  float bv[KNN]; int bi[KNN];
#pragma unroll
  for (int j = 0; j < KNN; ++j) { bv[j] = -1e30f; bi[j] = 0; }

  const int abase = (lane < 16) ? 0 : 8;
  for (int m0 = 0; m0 < NPTS; m0 += 16) {
    const bf16* rm = xb + (base + m0 + (lane & 15)) * 64;
    v16bf a0 = cat8(*(const v8bf*)(rm + kb), *(const v8bf*)(rm + kb + 8));
    v16bf a1 = cat8(*(const v8bf*)(rm + 32 + kb),
                    *(const v8bf*)(rm + 32 + kb + 8));
    v8f c = {};
    c = wmma_bf16(a0, bf0, c);
    c = wmma_bf16(a1, bf1, c);
#pragma unroll
    for (int v = 0; v < 8; ++v) {
      int m = m0 + abase + v;
      float d = 2.f * c[v] - xx[base + m] - xxq;
      if (d > bv[KNN - 1]) insert20(bv, bi, d, m);
    }
  }

  __shared__ float sv[32 * KNN];
  __shared__ int si[32 * KNN];
#pragma unroll
  for (int j = 0; j < KNN; ++j) {
    sv[lane * KNN + j] = bv[j];
    si[lane * KNN + j] = bi[j];
  }
  __syncthreads();
  if (lane < 16) {
    int ia = 0, ib = 0;
    long long orow = ((long long)b * NPTS + q0 + lane) * KNN;
    for (int j = 0; j < KNN; ++j) {
      float va = sv[lane * KNN + ia];
      float vb = sv[(lane + 16) * KNN + ib];
      int sel;
      if (va >= vb) { sel = si[lane * KNN + ia]; ++ia; }
      else          { sel = si[(lane + 16) * KNN + ib]; ++ib; }
      idx[orow + j] = sel;
    }
  }
}

// ---------------------------------------------------------------------------
// Host pipeline
// ---------------------------------------------------------------------------
extern "C" void kernel_launch(void* const* d_in, const int* in_sizes, int n_in,
                              void* d_out, int out_size, void* d_ws,
                              size_t ws_size, hipStream_t stream) {
  (void)in_sizes; (void)n_in; (void)out_size; (void)ws_size;
  const float* x = (const float*)d_in[0];
  const float* Wt[9] = {(const float*)d_in[1], (const float*)d_in[2],
                        (const float*)d_in[3], (const float*)d_in[4],
                        (const float*)d_in[5], (const float*)d_in[6],
                        (const float*)d_in[7], (const float*)d_in[8],
                        (const float*)d_in[9]};
  const float* g2d = (const float*)d_in[10];
  const float* b2d = (const float*)d_in[11];
  const float* g6 = (const float*)d_in[12];
  const float* b6 = (const float*)d_in[13];
  const float* g7 = (const float*)d_in[14];
  const float* b7 = (const float*)d_in[15];
  const float* g8 = (const float*)d_in[16];
  const float* b8 = (const float*)d_in[17];

  char* w = (char*)d_ws;
  auto alloc = [&](size_t bytes) -> char* {
    char* p = w;
    w += (bytes + 255) & ~(size_t)255;
    return p;
  };
  int* idx1 = (int*)alloc((size_t)BN * KNN * 4);
  int* idx2 = (int*)alloc((size_t)BN * KNN * 4);
  int* idx3 = (int*)alloc((size_t)BN * KNN * 4);
  bf16* xin = (bf16*)alloc((size_t)BN * 16 * 2);
  float* YA = (float*)alloc((size_t)64 * P1LL * 4);   // largest f32 tensor
  bf16* ACT = (bf16*)alloc((size_t)64 * P1LL * 2);    // largest bf16 tensor
  bf16* x1b = (bf16*)alloc((size_t)BN * 64 * 2);
  bf16* x2b = (bf16*)alloc((size_t)BN * 64 * 2);
  bf16* x3b = (bf16*)alloc((size_t)BN * 64 * 2);
  float* xx = (float*)alloc((size_t)BN * 4);
  float* mean = (float*)alloc(1024 * 4);
  float* var = (float*)alloc(1024 * 4);
  bf16* gm = (bf16*)alloc((size_t)1024 * BATCH * 2);
  bf16* WSW = (bf16*)alloc((size_t)2 * 1024 * 1024);  // pre-swizzled weights

  auto blks = [](long long n, int t) { return (unsigned)((n + t - 1) / t); };

  auto run_gemm = [&](int mode, const float* W, const bf16* X, const int* idx,
                      float* Y, int Cout, int Cin, long long P) {
    int Mt = (Cout + 15) / 16, nkt = (Cin + 31) / 32;
    long long nw = (long long)Mt * nkt * 512;
    prep_w<<<blks(nw, 256), 256, 0, stream>>>(W, WSW, Cout, Cin, Mt, nkt);
    dim3 g(Mt, (unsigned)(P / 64));
    if (mode == 0)
      gemm_wmma<0><<<g, 128, 0, stream>>>(WSW, X, nullptr, Y, Cout, Cin, P);
    else if (mode == 1)
      gemm_wmma<1><<<g, 128, 0, stream>>>(WSW, X, idx, Y, Cout, Cin, P);
    else
      gemm_wmma<2><<<g, 128, 0, stream>>>(WSW, X, idx, Y, Cout, Cin, P);
  };
  auto bn = [&](float* Y, long long P, int C, const float* g, const float* bb,
                bf16* out) {
    colstats<<<C, 256, 0, stream>>>(Y, P, mean, var);
    long long tot = (long long)C * P;
    bn_lrelu_bf16<<<blks(tot, 256), 256, 0, stream>>>(Y, mean, var, g, bb, out,
                                                      tot, P, C);
  };

  // ---- stage 0 ------------------------------------------------------------
  make_xin<<<blks((long long)BN * 16, 256), 256, 0, stream>>>(x, xin);
  knn3<<<BN / 256, 256, 0, stream>>>(x, idx1);

  // ---- edge block 1: W1 (18->64), W2 (64->64), max_k -> x1 ---------------
  run_gemm(2, Wt[0], xin, idx1, YA, 64, 18, P1LL);
  bn(YA, P1LL, 64, g2d + 0 * 64, b2d + 0 * 64, ACT);
  run_gemm(0, Wt[1], ACT, nullptr, YA, 64, 64, P1LL);
  bn(YA, P1LL, 64, g2d + 1 * 64, b2d + 1 * 64, ACT);
  kmax20<<<blks((long long)BN * 64, 256), 256, 0, stream>>>(ACT, x1b);

  // ---- kNN(x1) + edge block 2 --------------------------------------------
  xnorm64<<<BN / 256, 256, 0, stream>>>(x1b, xx);
  knn64<<<dim3(NPTS / 16, BATCH), 32, 0, stream>>>(x1b, xx, idx2);
  run_gemm(1, Wt[2], x1b, idx2, YA, 64, 128, P1LL);
  bn(YA, P1LL, 64, g2d + 2 * 64, b2d + 2 * 64, ACT);
  run_gemm(0, Wt[3], ACT, nullptr, YA, 64, 64, P1LL);
  bn(YA, P1LL, 64, g2d + 3 * 64, b2d + 3 * 64, ACT);
  kmax20<<<blks((long long)BN * 64, 256), 256, 0, stream>>>(ACT, x2b);

  // ---- kNN(x2) + edge block 3 --------------------------------------------
  xnorm64<<<BN / 256, 256, 0, stream>>>(x2b, xx);
  knn64<<<dim3(NPTS / 16, BATCH), 32, 0, stream>>>(x2b, xx, idx3);
  run_gemm(1, Wt[4], x2b, idx3, YA, 64, 128, P1LL);
  bn(YA, P1LL, 64, g2d + 4 * 64, b2d + 4 * 64, ACT);
  kmax20<<<blks((long long)BN * 64, 256), 256, 0, stream>>>(ACT, x3b);

  // ---- head: cat(x1,x2,x3) [BN][192] -> W6 -> global max -----------------
  copy_cols<<<blks((long long)BN * 64, 256), 256, 0, stream>>>(x1b, ACT, 192, 0, BN);
  copy_cols<<<blks((long long)BN * 64, 256), 256, 0, stream>>>(x2b, ACT, 192, 64, BN);
  copy_cols<<<blks((long long)BN * 64, 256), 256, 0, stream>>>(x3b, ACT, 192, 128, BN);
  run_gemm(0, Wt[5], ACT, nullptr, YA, 1024, 192, BN);
  bn(YA, BN, 1024, g6, b6, ACT);                       // e: [BN][1024]
  gmax_kernel<<<dim3(1024, BATCH), 256, 0, stream>>>(ACT, gm);

  // ---- cat(gmax,x1,x2,x3) [BN][1216] -> W7 -> W8 -> Wc -------------------
  bcast_gm<<<blks((long long)BN * 1024, 256), 256, 0, stream>>>(gm, ACT);
  copy_cols<<<blks((long long)BN * 64, 256), 256, 0, stream>>>(x1b, ACT, 1216, 1024, BN);
  copy_cols<<<blks((long long)BN * 64, 256), 256, 0, stream>>>(x2b, ACT, 1216, 1088, BN);
  copy_cols<<<blks((long long)BN * 64, 256), 256, 0, stream>>>(x3b, ACT, 1216, 1152, BN);
  run_gemm(0, Wt[6], ACT, nullptr, YA, 512, 1216, BN);
  bn(YA, BN, 512, g7, b7, ACT);
  run_gemm(0, Wt[7], ACT, nullptr, YA, 256, 512, BN);
  bn(YA, BN, 256, g8, b8, ACT);
  run_gemm(0, Wt[8], ACT, nullptr, YA, 17, 256, BN);
  out_xpose<<<blks((long long)17 * BN, 256), 256, 0, stream>>>(YA, (float*)d_out);
}
